// RNN_4535485464622
// MI455X (gfx1250) — matrix operand order repaired and checked by reading, not compiled, not run on hardware
//
#include <hip/hip_runtime.h>
#include <hip/hip_bf16.h>

// LSTM: B=128, T=512, D=1024, H=1024
#define B_  128
#define T_  512
#define D_  1024
#define H_  1024
#define G4H (4 * H_)

typedef __bf16 bf16_t;
typedef __attribute__((ext_vector_type(16))) __bf16 v16bf;
typedef __attribute__((ext_vector_type(8)))  __bf16 v8bf;
typedef __attribute__((ext_vector_type(8)))  float  v8f;
typedef __attribute__((ext_vector_type(4)))  int    v4i;

union ABfrag { v16bf v; v8bf h[2]; };

__device__ __forceinline__ float sigmoidf_(float x) {
    return 1.0f / (1.0f + __expf(-x));
}

// ---- CDNA5 async global->LDS staging (ASYNCcnt path), with safe fallback ----
#if defined(__HIP_DEVICE_COMPILE__) && \
    __has_builtin(__builtin_amdgcn_global_load_async_to_lds_b128)
#define ASYNC_LDS 1
#else
#define ASYNC_LDS 0
#endif

__device__ __forceinline__ void stage16(const bf16_t* g, bf16_t* l) {
#if ASYNC_LDS
    // Signature (from clang diagnostic): (AS1 v4i*, AS3 v4i*, imm offset, imm cpol)
    __builtin_amdgcn_global_load_async_to_lds_b128(
        (__attribute__((address_space(1))) v4i*)(g),
        (__attribute__((address_space(3))) v4i*)(l), 0, 0);
#else
    *(v8bf*)l = *(const v8bf*)g;
#endif
}

__device__ __forceinline__ void stage_wait() {
#if ASYNC_LDS
#if __has_builtin(__builtin_amdgcn_s_wait_asynccnt)
    __builtin_amdgcn_s_wait_asynccnt(0);
#else
    asm volatile("s_wait_asynccnt 0" ::: "memory");
#endif
#endif
}

// ---------------------------------------------------------------------------
// Prep: convert weights fp32 -> bf16 (row-major, same layout), zero h0 and c.
// ---------------------------------------------------------------------------
__global__ void lstm_prep(const float* __restrict__ Wih, const float* __restrict__ Whh,
                          bf16_t* __restrict__ wihb, bf16_t* __restrict__ whhb,
                          bf16_t* __restrict__ h0, float* __restrict__ c0) {
    const int nW = G4H * D_;      // 4,194,304
    const int nS = B_ * H_;       // 131,072
    for (int i = blockIdx.x * blockDim.x + threadIdx.x; i < nW;
         i += gridDim.x * blockDim.x) {
        wihb[i] = (bf16_t)Wih[i];
        whhb[i] = (bf16_t)Whh[i];
        if (i < nS) { h0[i] = (bf16_t)0.0f; c0[i] = 0.0f; }
    }
}

// ---------------------------------------------------------------------------
// One LSTM timestep. Block = 8 waves sharing one 16-wide hidden tile (ntile);
// wave w owns batch tile w. Block-shared W_hh/W_ih fragments are staged to
// LDS with async global->LDS loads (double-buffered, ASYNCcnt), WMMAs read
// B-fragments from LDS (ds_load_b128), A-fragments from global.
// Grid: 64 blocks x 256 threads = 512 waves = (128/16) * (1024/16) tiles.
// ---------------------------------------------------------------------------
__global__ __launch_bounds__(256) void lstm_step(
    const float*  __restrict__ x,       // [B, T, D] fp32 (batch-first!)
    const bf16_t* __restrict__ wih,     // [4H, D] bf16 row-major
    const bf16_t* __restrict__ whh,     // [4H, H] bf16 row-major
    const float*  __restrict__ bias,    // [4H]
    const bf16_t* __restrict__ h_prev,  // [B, H] bf16
    bf16_t*       __restrict__ h_next,  // [B, H] bf16
    float*        __restrict__ c,       // [B, H] fp32
    float*        __restrict__ out,     // [T, B, H] fp32
    int t)
{
    // Double-buffered weight staging: 2 bufs x 8 frags x (16 rows x 32 k) bf16 = 16 KB
    __shared__ __align__(16) bf16_t smem[2][8 * 16 * 32];

    const int lane  = threadIdx.x & 31;
    const int wave  = threadIdx.x >> 5;
    const int btile = wave << 4;               // batch tile base (0..112)
    const int ntile = blockIdx.x << 4;         // hidden tile base (0..1008)
    const int half  = lane >> 4;               // 0/1
    const int lm    = lane & 15;

    const int arow = btile + lm;               // A-matrix row (batch idx) per lane
    const int ncol = ntile + lm;               // B/C/D column (hidden idx) per lane

    // Stage one 8 KB weight chunk (4 gates x {whh,wih} x 16 rows x 32 k) into smem[buf].
    auto stage_chunk = [&](int kb, int buf) {
        const int tid = (int)threadIdx.x;
#pragma unroll
        for (int u2 = 0; u2 < 2; ++u2) {
            const int e   = tid * 2 + u2;      // 0..511 : 16-byte units
            const int f   = e >> 6;            // fragment 0..7 (q*2 + m)
            const int u   = e & 63;
            const int row = u >> 2;            // 0..15
            const int ko  = (u & 3) * 8;       // 0,8,16,24 (bf16 elems)
            const int q   = f >> 1;
            const bf16_t* W = (f & 1) ? wih : whh;
            const bf16_t* src =
                W + (size_t)(q * H_ + ntile + row) * 1024 + kb + ko;
            bf16_t* dst = &smem[buf][f * 512 + row * 32 + ko];
            stage16(src, dst);
        }
    };

    v8f acc[4];
#pragma unroll
    for (int q = 0; q < 4; ++q)
#pragma unroll
        for (int j = 0; j < 8; ++j) acc[q][j] = 0.0f;

    const bf16_t* hrow = h_prev + (size_t)arow * H_;
    const float*  xrow = x + ((size_t)arow * T_ + t) * D_;

    stage_chunk(0, 0);  // prologue

    for (int kb = 0; kb < H_; kb += 32) {
        const int buf = (kb >> 5) & 1;

        stage_wait();        // this wave's staging loads have landed in LDS
        __syncthreads();     // all waves' staging visible; prev buf free

        if (kb + 32 < H_) stage_chunk(kb + 32, buf ^ 1);

        // A fragment from h_prev (bf16, 16x32): lane holds K = half*16..+16
        ABfrag ah;
        ah.h[0] = *(const v8bf*)(hrow + kb + half * 16);
        ah.h[1] = *(const v8bf*)(hrow + kb + half * 16 + 8);

        // A fragment from x (fp32 -> bf16 on the fly)
        ABfrag ax;
        {
            const float* p0 = xrow + kb + half * 16;
            const float* p1 = xrow + kb + half * 16 + 8;
            v8bf t0, t1;
#pragma unroll
            for (int j = 0; j < 8; ++j) {
                t0[j] = (bf16_t)p0[j];
                t1[j] = (bf16_t)p1[j];
            }
            ax.h[0] = t0; ax.h[1] = t1;
        }

        if (kb + 32 < H_)
            __builtin_prefetch(xrow + kb + 32, 0, 1);   // global_prefetch_b8

#pragma unroll
        for (int q = 0; q < 4; ++q) {
            // B fragments from LDS (lane reads row lm, k = half*16..+16; buf toggles)
            const bf16_t* ph = &smem[buf][(q * 2 + 0) * 512 + lm * 32 + half * 16];
            const bf16_t* px = &smem[buf][(q * 2 + 1) * 512 + lm * 32 + half * 16];
            v16bf bh = *(const v16bf*)ph;
            v16bf bx = *(const v16bf*)px;
            acc[q] = __builtin_amdgcn_wmma_f32_16x16x32_bf16(
                false, ah.v, false, bh, (short)0, acc[q], false, false);
            acc[q] = __builtin_amdgcn_wmma_f32_16x16x32_bf16(
                false, ax.v, false, bx, (short)0, acc[q], false, false);
        }
    }

    // Epilogue: LSTM cell. C/D layout: vgpr r -> row = r + 8*half, col = lm.
    const float bi = bias[0 * H_ + ncol];
    const float bf = bias[1 * H_ + ncol];
    const float bg = bias[2 * H_ + ncol];
    const float bo = bias[3 * H_ + ncol];

#pragma unroll
    for (int r = 0; r < 8; ++r) {
        const int    brow = btile + half * 8 + r;
        const size_t idx  = (size_t)brow * H_ + ncol;
        const float gi = sigmoidf_(acc[0][r] + bi);
        const float gf = sigmoidf_(acc[1][r] + bf);
        const float gg = tanhf(acc[2][r] + bg);
        const float go = sigmoidf_(acc[3][r] + bo);
        const float cn = gf * c[idx] + gi * gg;
        c[idx] = cn;
        const float hn = go * tanhf(cn);
        out[((size_t)t * B_ + brow) * H_ + ncol] = hn;
        h_next[idx] = (bf16_t)hn;
    }
}

// ---------------------------------------------------------------------------
// Finalize: append h_last (= outputs[T-1]) and c_last to d_out.
// ---------------------------------------------------------------------------
__global__ void lstm_final(const float* __restrict__ out_last,
                           const float* __restrict__ c,
                           float* __restrict__ hdst, float* __restrict__ cdst) {
    const int i = blockIdx.x * blockDim.x + threadIdx.x;
    if (i < B_ * H_) {
        hdst[i] = out_last[i];
        cdst[i] = c[i];
    }
}

// ---------------------------------------------------------------------------
extern "C" void kernel_launch(void* const* d_in, const int* in_sizes, int n_in,
                              void* d_out, int out_size, void* d_ws, size_t ws_size,
                              hipStream_t stream) {
    const float* x    = (const float*)d_in[0];   // [B, T, D]
    const float* Wih  = (const float*)d_in[1];   // [4H, D]
    const float* Whh  = (const float*)d_in[2];   // [4H, H]
    const float* bias = (const float*)d_in[3];   // [4H]
    float* out = (float*)d_out;                  // [T,B,H] + [B,H] + [B,H]

    // Workspace layout (~17.8 MB total)
    char* ws = (char*)d_ws;
    bf16_t* wihb = (bf16_t*)(ws);                                    // 8 MB
    bf16_t* whhb = (bf16_t*)(ws + (size_t)8 * 1024 * 1024);          // 8 MB
    bf16_t* h0   = (bf16_t*)(ws + (size_t)16 * 1024 * 1024);         // 256 KB
    bf16_t* h1   = (bf16_t*)(ws + (size_t)16 * 1024 * 1024 + 256 * 1024); // 256 KB
    float*  cbuf = (float*) (ws + (size_t)16 * 1024 * 1024 + 512 * 1024); // 512 KB

    lstm_prep<<<1024, 256, 0, stream>>>(Wih, Whh, wihb, whhb, h0, cbuf);

    bf16_t* hb[2] = {h0, h1};
    for (int t = 0; t < T_; ++t) {
        lstm_step<<<64, 256, 0, stream>>>(x, wihb, whhb, bias,
                                          hb[t & 1], hb[(t + 1) & 1],
                                          cbuf, out, t);
    }

    const size_t nBH = (size_t)B_ * H_;
    lstm_final<<<(int)((nBH + 255) / 256), 256, 0, stream>>>(
        out + (size_t)(T_ - 1) * nBH, cbuf,
        out + (size_t)T_ * nBH, out + (size_t)T_ * nBH + nBH);
}
